// DifuscoGNNLayer_83528523972663
// MI455X (gfx1250) — matrix operand order repaired and checked by reading, not compiled, not run on hardware
//
#include <hip/hip_runtime.h>
#include <hip/hip_bf16.h>
#include <math.h>

typedef __attribute__((ext_vector_type(16))) __bf16 v16bf;
typedef __attribute__((ext_vector_type(8)))  __bf16 v8bf;
typedef __attribute__((ext_vector_type(8)))  float  v8f;

#define HDIM 128
#define FSTR 132   // f32 LDS tile row stride (528B: bank stride 4 mod 64 -> conflict-free)
#define BSTR 136   // bf16 LDS tile row stride (272B: bank stride 4 mod 64, 16B aligned)

// B-operand fragment (32x16 bf16, KxN): lane holds column n=lane&15; lane half
// selects the contiguous 16-element K chunk. B[k][n] = W[n][k] -> reads W rows.
__device__ __forceinline__ v16bf load_bfrag(const float* __restrict__ wr) {
  v16bf b;
#pragma unroll
  for (int p = 0; p < 16; ++p) b[p] = (__bf16)wr[p];
  return b;
}

// ---------------------------------------------------------------------------
// Kernel 1: node linears.  O_m = X @ W_m^T + b_m  for m in {U,V,A,B}.
// grid = (R/16, 4), block = 256 (8 waves); wave w computes n-tile [16w,16w+16).
// ---------------------------------------------------------------------------
__global__ __launch_bounds__(256) void difusco_node_gemm_kernel(
    const float* __restrict__ X,
    const float* __restrict__ W0, const float* __restrict__ W1,
    const float* __restrict__ W2, const float* __restrict__ W3,
    const float* __restrict__ B0, const float* __restrict__ B1,
    const float* __restrict__ B2, const float* __restrict__ B3,
    float* __restrict__ O0, float* __restrict__ O1,
    float* __restrict__ O2, float* __restrict__ O3) {
  __shared__ __bf16 btile[16 * BSTR];

  const int rt = blockIdx.x;
  const int m  = blockIdx.y;
  const float* W  = (m == 0) ? W0 : (m == 1) ? W1 : (m == 2) ? W2 : W3;
  const float* Bv = (m == 0) ? B0 : (m == 1) ? B1 : (m == 2) ? B2 : B3;
  float*       O  = (m == 0) ? O0 : (m == 1) ? O1 : (m == 2) ? O2 : O3;

  const int t    = threadIdx.x;
  const int wv   = t >> 5;
  const int lane = t & 31;
  const int l15  = lane & 15;
  const int hh   = lane >> 4;
  const int n    = wv * 16 + l15;

  {  // stage 16 rows x 128 of X into LDS, converted to bf16 once
    const int jr = t >> 4, c0 = (t & 15) * 8;
    const float* src = &X[(size_t)(rt * 16 + jr) * HDIM + c0];
    v8bf o;
#pragma unroll
    for (int p = 0; p < 8; ++p) o[p] = (__bf16)src[p];
    *(v8bf*)(&btile[jr * BSTR + c0]) = o;
  }
  __syncthreads();

  v8f acc = {};
#pragma unroll
  for (int s = 0; s < 4; ++s) {
    const int k0 = s * 32 + hh * 16;
    v8bf c1 = *(const v8bf*)(&btile[l15 * BSTR + k0]);
    v8bf c2 = *(const v8bf*)(&btile[l15 * BSTR + k0 + 8]);
    v16bf a;
#pragma unroll
    for (int p = 0; p < 8; ++p) { a[p] = c1[p]; a[p + 8] = c2[p]; }
    v16bf bf = load_bfrag(&W[(size_t)n * HDIM + s * 32 + hh * 16]);
    acc = __builtin_amdgcn_wmma_f32_16x16x32_bf16(
        false, a, false, bf, (short)0, acc, false, false);
  }

  const float bias = Bv[n];
#pragma unroll
  for (int r = 0; r < 8; ++r) {
    const int row = rt * 16 + r + hh * 8;  // D layout: M = r + 8*half
    O[(size_t)row * HDIM + n] = acc[r] + bias;
  }
}

// ---------------------------------------------------------------------------
// Kernel 2: fused edge pipeline. One block per (b,i) row; loops over N/16
// j-tiles with a double-buffered async-to-LDS pipeline for h_edges.
// Per tile: one-shot f32->bf16 convert in LDS, WMMA Ce, + Ah[i]+C_b+Bh[j],
// sigmoid gate * Vh[j] * mask -> agg registers, LayerNorm over H, ReLU,
// residual read from the f32 LDS tile. h_edges touches HBM exactly once.
// ---------------------------------------------------------------------------
__global__ __launch_bounds__(256) void difusco_edge_kernel(
    const float* __restrict__ h_edges, const float* __restrict__ adj,
    const float* __restrict__ Ah, const float* __restrict__ Bh,
    const float* __restrict__ Vh, const float* __restrict__ C_w,
    const float* __restrict__ C_b, const float* __restrict__ ge,
    const float* __restrict__ be, float* __restrict__ eout,
    float* __restrict__ agg, int N) {
  __shared__ float  fbuf[2][16 * FSTR];  // raw f32 tiles (async landing zone)
  __shared__ __bf16 bbuf[16 * BSTR];     // once-converted bf16 tile
  __shared__ float  red1[8][16];
  __shared__ float  red2[8][16];
  __shared__ float  mu_s[16];
  __shared__ float  rs_s[16];

  const int bi   = blockIdx.x;   // b*N + i
  const int t    = threadIdx.x;
  const int wv   = t >> 5;       // 0..7 -> n-tile
  const int lane = t & 31;
  const int l15  = lane & 15;
  const int hh   = lane >> 4;
  const int n    = wv * 16 + l15;
  const int b    = bi / N;

  // Loop-invariant B fragments of C_w (4 K-steps), kept in VGPRs.
  v16bf bfrag[4];
#pragma unroll
  for (int s = 0; s < 4; ++s)
    bfrag[s] = load_bfrag(&C_w[(size_t)n * HDIM + s * 32 + hh * 16]);

  const float addn = Ah[(size_t)bi * HDIM + n] + C_b[n];
  const float gen = ge[n], ben = be[n];
  float aggn = 0.0f;

  // Per-thread staging coordinates (fixed across jtiles): 32B per thread.
  const int srow = t >> 4;           // 0..15
  const int scol = (t & 15) * 8;     // 0..120
  const unsigned gbyte = (unsigned)((srow * HDIM + scol) * sizeof(float));
  const unsigned laddr0 = (unsigned)(size_t)(&fbuf[0][srow * FSTR + scol]);
  const unsigned laddr1 = (unsigned)(size_t)(&fbuf[1][srow * FSTR + scol]);

  // Async copy: 2 x b128 per thread; offset applies to both LDS and global.
  auto issue_tile = [&](int buf, int jbase) {
    const float* base = &h_edges[((size_t)bi * N + jbase) * HDIM];
    const unsigned laddr = buf ? laddr1 : laddr0;
    asm volatile(
        "global_load_async_to_lds_b128 %0, %1, %2\n\t"
        "global_load_async_to_lds_b128 %0, %1, %2 offset:16"
        :
        : "v"(laddr), "v"(gbyte), "s"(base)
        : "memory");
  };

  const int NT = N / 16;
  issue_tile(0, 0);
  asm volatile("s_wait_asynccnt 0x0" ::: "memory");
  __syncthreads();

  for (int jt = 0; jt < NT; ++jt) {
    const int cur   = jt & 1;
    const int jbase = jt * 16;
    if (jt + 1 < NT) issue_tile(1 - cur, jbase + 16);  // prefetch next tile

    {  // one-shot f32 -> bf16 conversion of current tile (8 elems/thread)
      const float* sp = &fbuf[cur][srow * FSTR + scol];
      v8bf o;
#pragma unroll
      for (int p = 0; p < 8; ++p) o[p] = (__bf16)sp[p];
      *(v8bf*)(&bbuf[srow * BSTR + scol]) = o;
    }
    __syncthreads();

    // Ce tile: K=128 in 4 WMMA steps; A fragments straight from bf16 LDS.
    v8f acc = {};
#pragma unroll
    for (int s = 0; s < 4; ++s) {
      const int k0 = s * 32 + hh * 16;  // A layout: lane half picks K chunk
      v8bf c1 = *(const v8bf*)(&bbuf[l15 * BSTR + k0]);
      v8bf c2 = *(const v8bf*)(&bbuf[l15 * BSTR + k0 + 8]);
      v16bf afrag;
#pragma unroll
      for (int p = 0; p < 8; ++p) { afrag[p] = c1[p]; afrag[p + 8] = c2[p]; }
      acc = __builtin_amdgcn_wmma_f32_16x16x32_bf16(
          false, afrag, false, bfrag[s], (short)0, acc, false, false);
    }

    float ev[8];
#pragma unroll
    for (int r = 0; r < 8; ++r) {
      const int j16 = r + hh * 8;        // D layout: M = r + 8*half
      const int j   = jbase + j16;
      const float bh = Bh[((size_t)b * N + j) * HDIM + n];
      const float e  = acc[r] + addn + bh;
      ev[r] = e;
      const float g  = 1.0f / (1.0f + __expf(-e));
      const float vh = Vh[((size_t)b * N + j) * HDIM + n];
      const float mk = adj[(size_t)bi * N + j];
      aggn += g * vh * mk;
      // per-j partial LN sums over this wave's 16 n's (xor stays in half)
      float p1 = e, p2 = e * e;
#pragma unroll
      for (int off = 8; off >= 1; off >>= 1) {
        p1 += __shfl_xor(p1, off, 32);
        p2 += __shfl_xor(p2, off, 32);
      }
      if (l15 == 0) { red1[wv][j16] = p1; red2[wv][j16] = p2; }
    }
    __syncthreads();
    if (t < 16) {  // finish LN stats across the 8 waves (full H=128)
      float a1 = 0.0f, a2 = 0.0f;
#pragma unroll
      for (int w = 0; w < 8; ++w) { a1 += red1[w][t]; a2 += red2[w][t]; }
      const float mu  = a1 * (1.0f / HDIM);
      const float var = a2 * (1.0f / HDIM) - mu * mu;
      mu_s[t] = mu;
      rs_s[t] = rsqrtf(var + 1e-5f);
    }
    __syncthreads();
#pragma unroll
    for (int r = 0; r < 8; ++r) {
      const int j16 = r + hh * 8;
      float y = (ev[r] - mu_s[j16]) * rs_s[j16] * gen + ben;
      y = fmaxf(y, 0.0f);
      // residual from f32 LDS tile; streaming (non-temporal) store of output
      const float res = fbuf[cur][j16 * FSTR + n] + y;
      __builtin_nontemporal_store(res,
          &eout[((size_t)bi * N + jbase + j16) * HDIM + n]);
    }
    // Ensure prefetched tile landed (per-wave) and all LDS readers are done
    // before the next iteration reuses the buffers.
    asm volatile("s_wait_asynccnt 0x0" ::: "memory");
    __syncthreads();
  }

  // lanes l and l+16 share the same n, cover complementary j halves
  aggn += __shfl_down(aggn, 16, 32);
  if (hh == 0) agg[(size_t)bi * HDIM + n] = aggn;
}

// ---------------------------------------------------------------------------
// Kernel 3: node update.  out = h_nodes + relu(LN(Uh + agg)).
// One 128-thread block per (b,i) row.
// ---------------------------------------------------------------------------
__global__ __launch_bounds__(128) void difusco_node_update_kernel(
    const float* __restrict__ h_nodes, const float* __restrict__ Uh,
    const float* __restrict__ agg, const float* __restrict__ gh,
    const float* __restrict__ bh, float* __restrict__ out) {
  __shared__ float sm1[4];
  __shared__ float sm2[4];
  const int row = blockIdx.x;
  const int h   = threadIdx.x;
  const size_t idx = (size_t)row * HDIM + h;
  const float x = Uh[idx] + agg[idx];
  float s1 = x, s2 = x * x;
#pragma unroll
  for (int off = 16; off >= 1; off >>= 1) {
    s1 += __shfl_xor(s1, off, 32);
    s2 += __shfl_xor(s2, off, 32);
  }
  if ((h & 31) == 0) { sm1[h >> 5] = s1; sm2[h >> 5] = s2; }
  __syncthreads();
  const float t1 = sm1[0] + sm1[1] + sm1[2] + sm1[3];
  const float t2 = sm2[0] + sm2[1] + sm2[2] + sm2[3];
  const float mu  = t1 * (1.0f / HDIM);
  const float var = t2 * (1.0f / HDIM) - mu * mu;
  float y = (x - mu) * rsqrtf(var + 1e-5f) * gh[h] + bh[h];
  y = fmaxf(y, 0.0f);
  out[idx] = h_nodes[idx] + y;
}

// ---------------------------------------------------------------------------
extern "C" void kernel_launch(void* const* d_in, const int* in_sizes, int n_in,
                              void* d_out, int out_size, void* d_ws,
                              size_t ws_size, hipStream_t stream) {
  (void)n_in; (void)out_size; (void)ws_size;
  const float* h_nodes = (const float*)d_in[0];
  const float* h_edges = (const float*)d_in[1];
  const float* adj     = (const float*)d_in[2];
  const float* U_w = (const float*)d_in[3];
  const float* U_b = (const float*)d_in[4];
  const float* V_w = (const float*)d_in[5];
  const float* V_b = (const float*)d_in[6];
  const float* A_w = (const float*)d_in[7];
  const float* A_b = (const float*)d_in[8];
  const float* B_w = (const float*)d_in[9];
  const float* B_b = (const float*)d_in[10];
  const float* C_w = (const float*)d_in[11];
  const float* C_b = (const float*)d_in[12];
  const float* gh  = (const float*)d_in[13];
  const float* bh  = (const float*)d_in[14];
  const float* ge  = (const float*)d_in[15];
  const float* be  = (const float*)d_in[16];

  const long long e0 = in_sizes[0];            // B*N*H
  const long long e2 = in_sizes[2];            // B*N*N
  const int N = (int)((e2 * HDIM) / e0);       // 384
  const int R = (int)(e0 / HDIM);              // B*N = 768

  float* nodes_out = (float*)d_out;
  float* edges_out = nodes_out + (size_t)R * HDIM;

  float* Uh  = (float*)d_ws;                   // ws: 5 * R*H floats (~1.9 MB)
  float* Vhp = Uh  + (size_t)R * HDIM;
  float* Ahp = Vhp + (size_t)R * HDIM;
  float* Bhp = Ahp + (size_t)R * HDIM;
  float* agg = Bhp + (size_t)R * HDIM;

  difusco_node_gemm_kernel<<<dim3(R / 16, 4), 256, 0, stream>>>(
      h_nodes, U_w, V_w, A_w, B_w, U_b, V_b, A_b, B_b, Uh, Vhp, Ahp, Bhp);
  difusco_edge_kernel<<<R, 256, 0, stream>>>(
      h_edges, adj, Ahp, Bhp, Vhp, C_w, C_b, ge, be, edges_out, agg, N);
  difusco_node_update_kernel<<<R, 128, 0, stream>>>(
      h_nodes, Uh, agg, gh, bh, nodes_out);
}
